// Conv2Demod_55576876810963
// MI455X (gfx1250) — matrix operand order repaired and checked by reading, not compiled, not run on hardware
//
#include <hip/hip_runtime.h>

// ---------------------------------------------------------------------------
// Conv2Demod for MI455X (gfx1250, wave32, WMMA + double-buffered async-to-LDS).
// Implicit-GEMM per batch: M=Cout=512, N=64x64, K=Cin*9=4608 (reordered as
// k' = (kh*3+kw)*512 + cin so each 64-wide K step has uniform (kh,kw)).
// bf16 x bf16 -> f32 via v_wmma_f32_16x16x32_bf16; A tiles streamed with
// global_load_async_to_lds_b128 overlapped against WMMA on the other buffer.
// ---------------------------------------------------------------------------

typedef __attribute__((ext_vector_type(16))) __bf16 v16bf;
typedef __attribute__((ext_vector_type(8)))  float  v8f;

#define BATCH 8
#define CIN   512
#define COUT  512
#define HW    64
#define KTOT  (CIN * 9)           // 4608
#define MTILE 256
#define NTILE 64                  // one output row
#define KB    64
#define NTILES (KTOT / KB)        // 72 K-tiles (even)

__device__ __forceinline__ unsigned short f2bf(float f) {
    union { float f; unsigned u; } v; v.f = f;
    unsigned r = v.u + 0x7FFFu + ((v.u >> 16) & 1u);   // round-to-nearest-even
    return (unsigned short)(r >> 16);
}

__device__ __forceinline__ void async_lds_b128(unsigned ldsOff, unsigned gOff,
                                               unsigned long long gBase) {
    // GVS form: mem = SGPR64 + VGPR32 offset; LDS dest addr per-lane in VGPR.
    asm volatile("global_load_async_to_lds_b128 %0, %1, %2"
                 :: "v"(ldsOff), "v"(gOff), "s"(gBase) : "memory");
}

__device__ __forceinline__ void wait_async0() {
#if __has_builtin(__builtin_amdgcn_s_wait_asynccnt)
    __builtin_amdgcn_s_wait_asynccnt(0);
#else
    asm volatile("s_wait_asynccnt 0x0" ::: "memory");
#endif
}

// ---------------- Kernel A: img fp32 -> bf16 -------------------------------
__global__ __launch_bounds__(256)
void cd_img_to_bf16(const float* __restrict__ img,
                    unsigned short* __restrict__ dst, int n4) {
    int idx = blockIdx.x * 256 + threadIdx.x;
    if (idx < n4) {
        float4 v = ((const float4*)img)[idx];
        ushort4 o;
        o.x = f2bf(v.x); o.y = f2bf(v.y); o.z = f2bf(v.z); o.w = f2bf(v.w);
        ((ushort4*)dst)[idx] = o;
    }
}

// ---------------- Kernel B: modulate + demodulate -> bf16, K-permuted ------
// grid = (COUT, BATCH). Output k' index = (kh*3+kw)*512 + i.
__global__ __launch_bounds__(256)
void cd_mod_weights(const float* __restrict__ weight,   // [O][I][3][3]
                    const float* __restrict__ s,        // [B][I]
                    unsigned short* __restrict__ wsA) { // [B][O][KTOT] bf16
    const int o = blockIdx.x;
    const int b = blockIdx.y;
    const int t = threadIdx.x;
    __shared__ float wbuf[KTOT];     // 18.4 KB
    __shared__ float red[256];

    const float* wrow = weight + (size_t)o * KTOT;
    const float* srow = s + (size_t)b * CIN;
    float acc = 0.f;
    for (int e = t; e < KTOT; e += 256) {
        int i = e / 9;
        float wv = wrow[e] * (srow[i] + 1.0f);
        wbuf[e] = wv;
        acc += wv * wv;
    }
    red[t] = acc;
    __syncthreads();
    #pragma unroll
    for (int st = 128; st > 0; st >>= 1) {
        if (t < st) red[t] += red[t + st];
        __syncthreads();
    }
    float d = rsqrtf(red[0] + 1e-8f);
    unsigned short* orow = wsA + ((size_t)b * COUT + o) * KTOT;
    for (int e = t; e < KTOT; e += 256) {
        int i = e / 9;
        int r = e - i * 9;                 // kh*3+kw
        orow[r * CIN + i] = f2bf(wbuf[e] * d);
    }
}

// ---------------- Kernel C: implicit-GEMM conv via WMMA --------------------
// grid = (COUT/MTILE, HW, BATCH); block = 256 threads = 8 waves (4M x 2N).
// Wave tile 64(M) x 32(N) = 4x2 accum tiles; K stepped by 64 (2 WMMA k-subs).
// Software-pipelined: async-load tile i+1 while WMMA'ing tile i.
__global__ __launch_bounds__(256)
void cd_conv_wmma(const unsigned short* __restrict__ wsA,    // [B][O][KTOT] bf16 (K-permuted)
                  const unsigned short* __restrict__ imgbf,  // [B][I][64][64] bf16
                  float* __restrict__ out) {                 // [B][O][64][64] f32
    __shared__ unsigned short ldsA[2][MTILE * KB];   // 2 x 32 KB  [m][k]
    __shared__ unsigned short ldsBT[2][NTILE * KB];  // 2 x  8 KB  [n][k] transposed

    const int t    = threadIdx.x;
    const int lane = t & 31;
    const int wave = t >> 5;
    const int wm   = wave & 3;        // 0..3 -> M offset (64 each)
    const int wn   = wave >> 2;       // 0..1 -> N offset (32 each)
    const int oBase = blockIdx.x * MTILE;
    const int ph    = blockIdx.y;     // output row
    const int b     = blockIdx.z;

    const unsigned short* Abase = wsA + ((size_t)b * COUT + oBase) * KTOT;
    const unsigned short* Ibase = imgbf + (size_t)b * CIN * HW * HW;
    const unsigned long long AbaseU = (unsigned long long)(uintptr_t)Abase;

    v8f acc[4][2];
    #pragma unroll
    for (int mi = 0; mi < 4; ++mi)
        #pragma unroll
        for (int ni = 0; ni < 2; ++ni)
            acc[mi][ni] = (v8f){0.f,0.f,0.f,0.f,0.f,0.f,0.f,0.f};

    // ---- loop-invariant cooperative-load decomposition ----
    // A: 256 rows x 64 k = 2048 b128 chunks -> 8 per thread (async to LDS).
    unsigned aChunk[8];   // byte offset within one A tile
    unsigned aGOff[8];    // byte offset within A slab (k-invariant part)
    #pragma unroll
    for (int c = 0; c < 8; ++c) {
        int chunk = t + c * 256;
        int row = chunk >> 3;             // 0..255
        int col = chunk & 7;              // 0..7 (8 bf16 each)
        aChunk[c] = (unsigned)((row * KB + col * 8) * 2);
        aGOff[c]  = (unsigned)(((size_t)row * KTOT + col * 8) * 2);
    }
    const unsigned ldsAaddr[2] = { (unsigned)(size_t)&ldsA[0][0],
                                   (unsigned)(size_t)&ldsA[1][0] };
    // B: 64 rows x 8 chunks = 512 tasks -> 2 per thread (scatter transpose).
    const int bKK[2] = { (t + 0)   >> 3, (t + 256) >> 3 };
    const int bJ [2] = { t & 7,          t & 7 };

    // ---- loop-invariant fragment LDS element offsets (ISA 7.12.2) ----
    const int l15 = lane & 15;
    const int hi  = lane >> 4;
    int aFragIdx[4], bFragIdx[2];
    #pragma unroll
    for (int mi = 0; mi < 4; ++mi)
        aFragIdx[mi] = (wm * 64 + mi * 16 + l15) * KB + (hi ? 8 : 0);
    #pragma unroll
    for (int ni = 0; ni < 2; ++ni)
        bFragIdx[ni] = (wn * 32 + ni * 16 + l15) * KB + (hi ? 8 : 0);

    // ---- tile loader: async A copy + im2col-transpose B scatter ----
    auto load_tiles = [&](int kStep, int buf) {
        const int r  = kStep >> 9;            // filter tap 0..8 (uniform)
        const int i0 = kStep & 511;           // channel base within tap
        const int kh = r / 3;
        const int kw = r - kh * 3;
        const int srcRow = ph + kh - 1;
        const bool rowOK = (srcRow >= 0) & (srcRow < HW);
        const unsigned gk = (unsigned)(kStep * 2);
        const unsigned ldsBaseA = ldsAaddr[buf];
        #pragma unroll
        for (int c = 0; c < 8; ++c)
            async_lds_b128(ldsBaseA + aChunk[c], aGOff[c] + gk, AbaseU);

        unsigned short* bufB = &ldsBT[buf][0];
        #pragma unroll
        for (int c = 0; c < 2; ++c) {
            const int kk = bKK[c];            // 0..63 channel within step
            const int j  = bJ[c];             // 8-px chunk
            union { uint4 q; unsigned short u[8]; } ld;
            if (rowOK)
                ld.q = *(const uint4*)(Ibase + ((size_t)(i0 + kk) * HW + srcRow) * HW + j * 8);
            else
                ld.q = make_uint4(0u, 0u, 0u, 0u);
            #pragma unroll
            for (int q = 0; q < 8; ++q) {
                int n = j * 8 + q - kw + 1;   // B[k][n] = img[..][n+kw-1]
                if (n >= 0 && n < NTILE) bufB[n * KB + kk] = ld.u[q];
            }
            if (j == 0 && kw == 0) bufB[0  * KB + kk] = 0;   // x=-1 pad
            if (j == 7 && kw == 2) bufB[63 * KB + kk] = 0;   // x=64 pad
        }
    };

    // ---- tile compute: 2 k-substeps x (4M x 2N) = 16 WMMAs per wave ----
    auto compute_tile = [&](int buf) {
        const unsigned short* bufA = &ldsA[buf][0];
        const unsigned short* bufB = &ldsBT[buf][0];
        union Frag { v16bf v; uint4 q[2]; };
        #pragma unroll
        for (int ks = 0; ks < 2; ++ks) {
            Frag afrag[4], bfrag[2];
            #pragma unroll
            for (int mi = 0; mi < 4; ++mi) {
                afrag[mi].q[0] = *(const uint4*)(bufA + aFragIdx[mi] + ks * 32);
                afrag[mi].q[1] = *(const uint4*)(bufA + aFragIdx[mi] + ks * 32 + 16);
            }
            #pragma unroll
            for (int ni = 0; ni < 2; ++ni) {
                bfrag[ni].q[0] = *(const uint4*)(bufB + bFragIdx[ni] + ks * 32);
                bfrag[ni].q[1] = *(const uint4*)(bufB + bFragIdx[ni] + ks * 32 + 16);
            }
            #pragma unroll
            for (int mi = 0; mi < 4; ++mi)
                #pragma unroll
                for (int ni = 0; ni < 2; ++ni)
                    acc[mi][ni] = __builtin_amdgcn_wmma_f32_16x16x32_bf16(
                        false, afrag[mi].v, false, bfrag[ni].v,
                        (short)0, acc[mi][ni], false, false);
        }
    };

    // ---- software pipeline: prefetch next tile while computing current ----
    load_tiles(0, 0);
    wait_async0();
    __syncthreads();
    for (int it = 0; it < NTILES; it += 2) {
        load_tiles((it + 1) * KB, 1);         // it+1 <= 71 always
        compute_tile(0);
        wait_async0();
        __syncthreads();
        if (it + 2 < NTILES) load_tiles((it + 2) * KB, 0);
        compute_tile(1);
        wait_async0();
        __syncthreads();
    }

    // ---- store: C layout VGPR r -> M = r + 8*(lane>=16), N = lane&15 ----
    #pragma unroll
    for (int mi = 0; mi < 4; ++mi) {
        #pragma unroll
        for (int ni = 0; ni < 2; ++ni) {
            int nOut = wn * 32 + ni * 16 + l15;
            #pragma unroll
            for (int rr = 0; rr < 8; ++rr) {
                int mOut = wm * 64 + mi * 16 + rr + (hi << 3);
                out[(((size_t)b * COUT + oBase + mOut) * HW + ph) * HW + nOut] =
                    acc[mi][ni][rr];
            }
        }
    }
}

// ---------------------------------------------------------------------------
extern "C" void kernel_launch(void* const* d_in, const int* in_sizes, int n_in,
                              void* d_out, int out_size, void* d_ws, size_t ws_size,
                              hipStream_t stream) {
    const float* img    = (const float*)d_in[0];   // [8][512][64][64]
    const float* s      = (const float*)d_in[1];   // [8][512]
    const float* weight = (const float*)d_in[2];   // [512][512][3][3]
    float* out = (float*)d_out;

    // workspace: bf16 weights [8][512][4608] (37.75MB) + bf16 img (33.5MB)
    unsigned short* wsA   = (unsigned short*)d_ws;
    unsigned short* imgbf = wsA + (size_t)BATCH * COUT * KTOT;

    int n4 = BATCH * CIN * HW * HW / 4;    // 4,194,304 float4 groups
    cd_img_to_bf16<<<n4 / 256, 256, 0, stream>>>(img, imgbf, n4);

    cd_mod_weights<<<dim3(COUT, BATCH), 256, 0, stream>>>(weight, s, wsA);

    cd_conv_wmma<<<dim3(COUT / MTILE, HW, BATCH), 256, 0, stream>>>(wsA, imgbf, out);
}
